// VGG_SMALL_1w1aChannel_79663053406372
// MI455X (gfx1250) — 8-bit matrix operand order repaired and checked by reading, compiled, not run on hardware
//
#include <hip/hip_runtime.h>
#include <stdint.h>


typedef __attribute__((ext_vector_type(16))) int   v16i;
typedef __attribute__((ext_vector_type(8)))  float v8f;

#define FP8_P1 0x38
#define FP8_M1 0xB8

struct alignas(8)  U2 { uint32_t x, y; };
struct alignas(16) U4 { uint32_t x, y, z, w; };

__global__ void k_zero(uint32_t* __restrict__ p, size_t n32) {
  for (size_t i = (size_t)blockIdx.x * blockDim.x + threadIdx.x; i < n32;
       i += (size_t)gridDim.x * blockDim.x)
    p[i] = 0u;
}

__global__ void k_pack_input(const float* __restrict__ x, uint8_t* __restrict__ act,
                             int sY, int sB) {
  int i = blockIdx.x * 256 + threadIdx.x;
  if (i >= 256 * 3 * 1024) return;
  int xx = i & 31, yy = (i >> 5) & 31;
  int q = i >> 10;
  int b = q / 3, c = q - b * 3;
  act[(size_t)b * sB + (size_t)(yy + 1) * sY + (size_t)(xx + 1) * 16 + c] =
      (x[i] >= 0.f) ? (uint8_t)FP8_P1 : (uint8_t)FP8_M1;
}

__global__ void k_pack_w(const float* __restrict__ w, uint8_t* __restrict__ wpk,
                         float* __restrict__ alpha, int Cin, int cs, int Kpad) {
  int o = blockIdx.x, t = threadIdx.x;
  const float* wr = w + (size_t)o * Cin * 9;
  uint32_t* row32 = (uint32_t*)(wpk + (size_t)o * Kpad);
  for (int k = t; k < Kpad / 4; k += 256) row32[k] = 0u;
  __shared__ float red[256];
  int KW = Cin * 9;
  float s = 0.f;
  for (int k = t; k < KW; k += 256) s += fabsf(wr[k]);
  red[t] = s;
  __syncthreads();
  for (int off = 128; off > 0; off >>= 1) {
    if (t < off) red[t] += red[t + off];
    __syncthreads();
  }
  if (t == 0) alpha[o] = red[0] / (float)KW;
  uint8_t* row = wpk + (size_t)o * Kpad;
  for (int k = t; k < KW; k += 256) {
    int ci = k / 9, tap = k - ci * 9;
    row[tap * cs + ci] = (wr[k] >= 0.f) ? (uint8_t)FP8_P1 : (uint8_t)FP8_M1;
  }
}

template <bool SINGLE_TAP>
__global__ __launch_bounds__(256) void k_conv_gemm(
    const uint8_t* __restrict__ act, const uint8_t* __restrict__ wpk,
    const float* __restrict__ alpha, float* __restrict__ out,
    int N, int Kpad, int Ktot, int nslices,
    int log2cs, int Wlog, int HWlog, int sY, int sB) {
  const int tid   = threadIdx.x;
  const int lane  = tid & 31;
  const int wid   = tid >> 5;
  const int lr    = lane & 15;
  const int khalf = lane >> 4;
  const int m0 = blockIdx.x * 128 + (wid & 1) * 64;
  const int n0 = blockIdx.y * 128 + (wid >> 1) * 32;
  const int W  = 1 << Wlog;
  const int cs = 1 << log2cs;

  int pixbase[4];
#pragma unroll
  for (int mi = 0; mi < 4; ++mi) {
    int m   = m0 + mi * 16 + lr;
    int b   = m >> HWlog;
    int rem = m & ((1 << HWlog) - 1);
    int y   = rem >> Wlog;
    int x   = rem & (W - 1);
    pixbase[mi] = b * sB + y * sY + x * cs + khalf * 8;
  }

  const uint8_t* wr0 = wpk + (size_t)(n0 + lr) * Kpad + khalf * 16;
  const uint8_t* wr1 = wr0 + (size_t)16 * Kpad;

  const int tapShift = log2cs - 7;
  const int tapMask  = (1 << (tapShift > 0 ? tapShift : 0)) - 1;

  v8f acc[4][2] = {};

  for (int s = 0; s < nslices; ++s) {
    union BFrag { v16i v; U4 q[4]; } bf[2];
#pragma unroll
    for (int g = 0; g < 4; ++g) bf[0].q[g] = *(const U4*)(wr0 + g * 32);
#pragma unroll
    for (int g = 0; g < 4; ++g) bf[1].q[g] = *(const U4*)(wr1 + g * 32);
    wr0 += 128;
    wr1 += 128;

    int soff = 0;
    if (SINGLE_TAP) {
      const int tap = s >> tapShift;
      const int kh = tap / 3, kw = tap - kh * 3;
      soff = kh * sY + kw * cs + ((s & tapMask) << 7);
    }
    const int kbase_s = s * 128;

#pragma unroll
    for (int mi = 0; mi < 4; ++mi) {
      union AFrag { v16i v; U2 d[8]; } af;
      if (SINGLE_TAP) {
        const uint8_t* pa = act + (pixbase[mi] + soff);
#pragma unroll
        for (int blk = 0; blk < 2; ++blk)
#pragma unroll
          for (int p = 0; p < 4; ++p)
            af.d[blk * 4 + p] = *(const U2*)(pa + blk * 64 + (p >> 1) * 32 + (p & 1) * 8 + khalf * 8);
      } else {
#pragma unroll
        for (int blk = 0; blk < 2; ++blk) {
#pragma unroll
          for (int p = 0; p < 4; ++p) {
            const int kl = blk * 64 + (p >> 1) * 32 + (p & 1) * 8 + khalf * 16;
            const int kg = kbase_s + kl;
            U2 val; val.x = 0u; val.y = 0u;
            if (kg < Ktot) {
              const int tap  = kg >> log2cs;
              const int coff = kg & (cs - 1);
              const int kh = tap / 3, kw = tap - kh * 3;
              val = *(const U2*)(act + (pixbase[mi] - khalf * 8) + kh * sY +
                                 kw * cs + coff);
            }
            af.d[blk * 4 + p] = val;
          }
        }
      }
#pragma unroll
      for (int ni = 0; ni < 2; ++ni)
        acc[mi][ni] = __builtin_amdgcn_wmma_f32_16x16x128_fp8_fp8(
            af.v, bf[ni].v, (short)0, acc[mi][ni], false, false);
    }
  }

#pragma unroll
  for (int ni = 0; ni < 2; ++ni) {
    int n = n0 + ni * 16 + lr;
    float al = alpha[n];
#pragma unroll
    for (int mi = 0; mi < 4; ++mi) {
#pragma unroll
      for (int r = 0; r < 8; ++r) {
        int m = m0 + mi * 16 + r + khalf * 8;
        out[(size_t)m * N + n] = al * acc[mi][ni][r];
      }
    }
  }
}

__global__ void k_pool(const float* __restrict__ in, float* __restrict__ out,
                       int logC, int logW2, int logHW2, int total) {
  int i = blockIdx.x * 256 + threadIdx.x;
  if (i >= total) return;
  int C = 1 << logC, W2 = 1 << logW2;
  int c  = i & (C - 1);
  int p  = i >> logC;
  int x2 = p & (W2 - 1);
  int y2 = (p >> logW2) & ((1 << (logHW2 - logW2)) - 1);
  int b  = p >> logHW2;
  int Wf = W2 * 2;
  int HWin = 1 << (logHW2 + 2);
  size_t base = ((size_t)b * HWin + (size_t)(2 * y2) * Wf + 2 * x2) * C + c;
  size_t rs = (size_t)Wf * C;
  float v = fmaxf(fmaxf(in[base], in[base + C]),
                  fmaxf(in[base + rs], in[base + rs + C]));
  out[i] = v;
}

__global__ void k_stats(const float* __restrict__ x, float* __restrict__ sums,
                        int logC, size_t total) {
  int C = 1 << logC;
  __shared__ float s1[512], s2[512];
  for (int c = threadIdx.x; c < C; c += 256) { s1[c] = 0.f; s2[c] = 0.f; }
  __syncthreads();
  for (size_t i = (size_t)blockIdx.x * 256 + threadIdx.x; i < total;
       i += (size_t)gridDim.x * 256) {
    float v = x[i];
    int c = (int)(i & (size_t)(C - 1));
    atomicAdd(&s1[c], v);
    atomicAdd(&s2[c], v * v);
  }
  __syncthreads();
  for (int c = threadIdx.x; c < C; c += 256) {
    atomicAdd(&sums[c], s1[c]);
    atomicAdd(&sums[C + c], s2[c]);
  }
}

__global__ void k_bnfin(const float* __restrict__ sums,
                        const float* __restrict__ gamma,
                        const float* __restrict__ beta,
                        float* __restrict__ coef, int C, float invP) {
  int c = blockIdx.x * 256 + threadIdx.x;
  if (c >= C) return;
  float mean = sums[c] * invP;
  float var  = sums[C + c] * invP - mean * mean;
  float a    = gamma[c] * rsqrtf(var + 1e-5f);
  coef[c]     = a;
  coef[C + c] = beta[c] - mean * a;
}

__global__ void k_binact(const float* __restrict__ x, const float* __restrict__ coef,
                         uint8_t* __restrict__ act, int logC, int logW, int logHW,
                         int sY, int sB, int total) {
  int i = blockIdx.x * 256 + threadIdx.x;
  if (i >= total) return;
  int C  = 1 << logC;
  int c  = i & (C - 1);
  int p  = i >> logC;
  int xx = p & ((1 << logW) - 1);
  int yy = (p >> logW) & ((1 << (logHW - logW)) - 1);
  int b  = p >> logHW;
  float v = x[i] * coef[c] + coef[C + c];
  act[(size_t)b * sB + (size_t)(yy + 1) * sY + (size_t)(xx + 1) * C + c] =
      (v >= 0.f) ? (uint8_t)FP8_P1 : (uint8_t)FP8_M1;
}

__global__ void k_bnout(const float* __restrict__ x, const float* __restrict__ coef,
                        float* __restrict__ out, int logC, int logHW, int total) {
  int i = blockIdx.x * 256 + threadIdx.x;
  if (i >= total) return;
  int C   = 1 << logC;
  int c   = i & (C - 1);
  int p   = i >> logC;
  int b   = p >> logHW;
  int pix = p & ((1 << logHW) - 1);
  int HW  = 1 << logHW;
  out[(size_t)b * C * HW + (size_t)c * HW + pix] = x[i] * coef[c] + coef[C + c];
}

__global__ void k_fc(const float* __restrict__ h, const float* __restrict__ w,
                     const float* __restrict__ bias, float* __restrict__ out) {
  int b = blockIdx.x, t = threadIdx.x;
  float p[10];
#pragma unroll
  for (int j = 0; j < 10; ++j) p[j] = 0.f;
  const float* hb = h + (size_t)b * 8192;
  for (int k = t; k < 8192; k += 256) {
    float hv = hb[k];
#pragma unroll
    for (int j = 0; j < 10; ++j) p[j] += hv * w[j * 8192 + k];
  }
  __shared__ float red[256];
  for (int j = 0; j < 10; ++j) {
    red[t] = p[j];
    __syncthreads();
    for (int off = 128; off > 0; off >>= 1) {
      if (t < off) red[t] += red[t + off];
      __syncthreads();
    }
    if (t == 0) out[b * 10 + j] = red[0] + bias[j];
    __syncthreads();
  }
}

extern "C" void kernel_launch(void* const* d_in, const int* in_sizes, int n_in,
                              void* d_out, int out_size, void* d_ws, size_t ws_size,
                              hipStream_t stream) {
  (void)in_sizes; (void)n_in; (void)out_size; (void)ws_size;
  const float* x = (const float*)d_in[0];
  const float* convw[6]; const float* gam[6]; const float* bet[6];
  for (int i = 0; i < 6; ++i) {
    convw[i] = (const float*)d_in[1 + i];
    gam[i]   = (const float*)d_in[7 + i];
    bet[i]   = (const float*)d_in[13 + i];
  }
  const float* fcw = (const float*)d_in[19];
  const float* fcb = (const float*)d_in[20];

  uint8_t* ws = (uint8_t*)d_ws;
  size_t off = 0;
  auto alloc = [&](size_t bytes) -> void* {
    void* p = ws + off;
    off += (bytes + 255) & ~(size_t)255;
    return p;
  };
  float*   bufConv  = (float*)alloc(134217728);
  float*   bufPool  = (float*)alloc(33554432);
  uint8_t* bufAct   = (uint8_t*)alloc(38000000);
  float*   bufH     = (float*)alloc(8388608);
  uint8_t* bufW     = (uint8_t*)alloc(2359296);
  float*   bufAlpha = (float*)alloc(2048);
  float*   bufSums  = (float*)alloc(4096);
  float*   bufCoef  = (float*)alloc(4096);

  const int Cin[6]  = {3, 128, 128, 256, 256, 512};
  const int csA[6]  = {16, 128, 128, 256, 256, 512};
  const int Cout[6] = {128, 128, 256, 256, 512, 512};
  const int Hh[6]   = {32, 32, 16, 16, 8, 8};
  const int pool[6] = {0, 1, 0, 1, 0, 1};

  {
    size_t actBytes = (size_t)256 * 34 * 34 * 16;
    k_zero<<<1024, 256, 0, stream>>>((uint32_t*)bufAct, actBytes / 4);
    k_pack_input<<<(256 * 3 * 1024 + 255) / 256, 256, 0, stream>>>(
        x, bufAct, 34 * 16, 34 * 34 * 16);
  }

  for (int L = 0; L < 6; ++L) {
    const int H = Hh[L], W = H, M = 256 * H * W, N = Cout[L];
    const int cs = csA[L];
    const int Ktot = 9 * cs;
    const int Kpad = (Ktot + 127) & ~127;
    const int nsl  = Kpad / 128;
    const int log2cs = __builtin_ctz(cs);
    const int Wlog = __builtin_ctz(W), HWlog = 2 * Wlog;
    const int sY = (W + 2) * cs, sB = (H + 2) * (W + 2) * cs;

    k_pack_w<<<N, 256, 0, stream>>>(convw[L], bufW, bufAlpha, Cin[L], cs, Kpad);

    dim3 grid(M / 128, N / 128);
    if (L == 0) {
      k_conv_gemm<false><<<grid, 256, 0, stream>>>(
          bufAct, bufW, bufAlpha, bufConv, N, Kpad, Ktot, nsl, log2cs, Wlog,
          HWlog, sY, sB);
    } else {
      k_conv_gemm<true><<<grid, 256, 0, stream>>>(
          bufAct, bufW, bufAlpha, bufConv, N, Kpad, Ktot, nsl, log2cs, Wlog,
          HWlog, sY, sB);
    }

    const float* src = bufConv;
    int Ho = H, Wo = W;
    if (pool[L]) {
      Ho = H / 2; Wo = W / 2;
      int total = 256 * Ho * Wo * N;
      k_pool<<<(total + 255) / 256, 256, 0, stream>>>(
          bufConv, bufPool, __builtin_ctz(N), __builtin_ctz(Wo),
          __builtin_ctz(Ho * Wo), total);
      src = bufPool;
    }

    const size_t P = (size_t)256 * Ho * Wo;
    const size_t totalE = P * N;
    const int logC = __builtin_ctz(N);
    k_zero<<<1, 256, 0, stream>>>((uint32_t*)bufSums, (size_t)2 * N);
    k_stats<<<1024, 256, 0, stream>>>(src, bufSums, logC, totalE);
    k_bnfin<<<(N + 255) / 256, 256, 0, stream>>>(bufSums, gam[L], bet[L],
                                                 bufCoef, N, 1.0f / (float)P);

    if (L < 5) {
      const int sYn = (Wo + 2) * N, sBn = (Ho + 2) * (Wo + 2) * N;
      const size_t actBytes = (size_t)256 * sBn;
      k_zero<<<2048, 256, 0, stream>>>((uint32_t*)bufAct, actBytes / 4);
      k_binact<<<((int)totalE + 255) / 256, 256, 0, stream>>>(
          src, bufCoef, bufAct, logC, __builtin_ctz(Wo),
          __builtin_ctz(Ho * Wo), sYn, sBn, (int)totalE);
    } else {
      k_bnout<<<((int)totalE + 255) / 256, 256, 0, stream>>>(
          src, bufCoef, bufH, logC, __builtin_ctz(Ho * Wo), (int)totalE);
    }
  }

  k_fc<<<256, 256, 0, stream>>>(bufH, fcw, fcb, (float*)d_out);
}
